// Net_coor_cent_85478439125046
// MI455X (gfx1250) — compile-verified, matrix operand order checked by reading, not run on hardware
//
#include <hip/hip_runtime.h>
#include <hip/hip_bf16.h>

// ---------------------------------------------------------------------------
// CDNA5 (gfx1250) implementation of a 4-layer PyG TransformerConv GNN.
//  - Dense projections: v_wmma_f32_16x16x32_bf16, bf16 operands pre-packed,
//    B tiles staged into LDS by the Tensor Data Mover (tensor_load_to_lds).
//  - Edge softmax/aggregation: L2-resident gathers + f32 atomics.
// ---------------------------------------------------------------------------

typedef __attribute__((ext_vector_type(16))) __bf16        v16bf;
typedef __attribute__((ext_vector_type(8)))  float         v8f;
typedef __attribute__((ext_vector_type(4)))  unsigned int  v4u;
typedef __attribute__((ext_vector_type(8)))  int           v8i_t;
typedef __attribute__((ext_vector_type(4)))  int           v4i_t;

#if defined(__has_builtin)
#if __has_builtin(__builtin_amdgcn_tensor_load_to_lds) && \
    __has_builtin(__builtin_amdgcn_s_wait_tensorcnt)
#define USE_TDM 1
#endif
#endif

__device__ __forceinline__ unsigned short f2bfbits(float f) {
  unsigned int u = __float_as_uint(f);
  return (unsigned short)((u + 0x7FFFu + ((u >> 16) & 1u)) >> 16);  // RNE
}

// ---------------------------------------------------------------------------
// Operand preparation
// ---------------------------------------------------------------------------
__global__ void cvt_f32_bf16(const float* __restrict__ in,
                             unsigned short* __restrict__ out, long n) {
  long i = blockIdx.x * (long)blockDim.x + threadIdx.x;
  const long st = (long)gridDim.x * blockDim.x;
  for (; i < n; i += st) out[i] = f2bfbits(in[i]);
}

// Pack W (indexed W[k*sK + n*sN], K x Nc) into B-fragment dwords:
//   Bpk[kk*Nc + n] = { bf16(W[2kk, n]), bf16(W[2kk+1, n]) }
__global__ void pack_B(const float* __restrict__ W, int sK, int sN,
                       unsigned int* __restrict__ Bpk, int K, int Nc) {
  const int total = (K >> 1) * Nc;
  int i = (int)(blockIdx.x * blockDim.x + threadIdx.x);
  if (i >= total) return;
  const int kk = i / Nc, n = i % Nc;
  const unsigned int lo = f2bfbits(W[(size_t)(2 * kk) * sK + (size_t)n * sN]);
  const unsigned int hi = f2bfbits(W[(size_t)(2 * kk + 1) * sK + (size_t)n * sN]);
  Bpk[i] = lo | (hi << 16);
}

// ---------------------------------------------------------------------------
// WMMA GEMM: C[M,Nc] = A[M,K](bf16) * Bpk(+bias).
// Each wave computes NT adjacent 16x16 tiles (A fragment reused),
// 8 waves (128 rows) per block, B tile staged in LDS via TDM.
// ---------------------------------------------------------------------------
template <int K, int NT>
__global__ void wmma_gemm_bf16(const unsigned short* __restrict__ Abf,
                               const unsigned int* __restrict__ Bpk, int Nc,
                               const float* __restrict__ bias,
                               float* __restrict__ C,
                               unsigned short* __restrict__ Cbf,  // optional
                               int M)
{
  constexpr int KK = K >> 1;                 // Bpk rows
  constexpr int TW = 16 * NT;                // staged tile width (dwords)
  __shared__ unsigned int ldsB[KK * TW];
  const int lane = (int)(threadIdx.x & 31);
  const int wid  = (int)(threadIdx.x >> 5);
  const int ng   = (int)blockIdx.y;          // n-tile group

  // ---- stage B tile (KK x TW dwords, row stride Nc) into LDS ----
#ifdef USE_TDM
  if (threadIdx.x < 32) {
    const unsigned long long ga =
        (unsigned long long)(const void*)(Bpk + (size_t)ng * TW);
    const unsigned int lds_base =
        (unsigned int)(unsigned long long)(void*)&ldsB[0];
    v4u g0;
    g0[0] = 1u;                                           // count=1
    g0[1] = lds_base;                                     // lds_addr
    g0[2] = (unsigned int)(ga & 0xFFFFFFFFu);             // global_addr lo
    g0[3] = (unsigned int)((ga >> 32) & 0x1FFFFFFu) | (2u << 30);  // hi|type=2
    v8i_t g1;
    g1[0] = (int)(2u << 16);                              // data_size = 4B
    g1[1] = (int)(((unsigned)Nc & 0xFFFFu) << 16);        // tensor_dim0 lo16
    g1[2] = (int)((((unsigned)KK & 0xFFFFu) << 16) |      // tensor_dim1 lo16
                  (((unsigned)Nc >> 16) & 0xFFFFu));      // | tensor_dim0 hi16
    g1[3] = (int)((unsigned)TW << 16);                    // tile_dim0 = TW
    g1[4] = (int)((unsigned)KK & 0xFFFFu);                // tile_dim1 = K/2
    g1[5] = (int)(unsigned)Nc;                            // dim0_stride lo32
    g1[6] = 0;
    g1[7] = 0;
    v4i_t gz = {0, 0, 0, 0};
#if __clang_major__ >= 23
    v8i_t gz8 = {0, 0, 0, 0, 0, 0, 0, 0};
    __builtin_amdgcn_tensor_load_to_lds(g0, g1, gz, gz, gz8, 0);
#else
    __builtin_amdgcn_tensor_load_to_lds(g0, g1, gz, gz, 0);
#endif
    __builtin_amdgcn_s_wait_tensorcnt(0);
  }
#else
  for (int idx = (int)threadIdx.x; idx < KK * TW; idx += 256)
    ldsB[idx] = Bpk[(size_t)(idx / TW) * Nc + ng * TW + (idx % TW)];
#endif
  __syncthreads();

  const int mtile = (int)blockIdx.x * 8 + wid;
  if (mtile * 16 >= M) return;               // wave-uniform exit (after barrier)

  const int r = lane & 15;
  const int half = lane >> 4;
  int arow = mtile * 16 + r;
  if (arow >= M) arow = M - 1;               // clamp; stores guarded below
  const uint4* Arow = (const uint4*)(Abf + (size_t)arow * K);

  const v8f vzero = {0.f, 0.f, 0.f, 0.f, 0.f, 0.f, 0.f, 0.f};
  v8f acc[NT];
#pragma unroll
  for (int t = 0; t < NT; ++t) acc[t] = vzero;

#pragma unroll
  for (int kb = 0; kb < K; kb += 32) {
    // A fragment: two contiguous 16B chunks per lane (ISA 16-bit A layout)
    union { uint4 q[2]; v16bf v; } ua;
    ua.q[0] = Arow[(kb >> 3) + half];
    ua.q[1] = Arow[(kb >> 3) + half + 2];
    const int base = (kb >> 1) + (half << 2);
#pragma unroll
    for (int t = 0; t < NT; ++t) {
      union { unsigned int u[8]; v16bf v; } ub;
#pragma unroll
      for (int i = 0; i < 8; ++i) {
        const int kk = base + ((i & 4) << 1) + (i & 3);
        ub.u[i] = ldsB[kk * TW + t * 16 + r];
      }
      acc[t] = __builtin_amdgcn_wmma_f32_16x16x32_bf16(
          false, ua.v, false, ub.v, (short)0, acc[t], false, false);
    }
  }

  const bool full = (mtile * 16 + 16) <= M;  // wave-uniform
#pragma unroll
  for (int t = 0; t < NT; ++t) {
    const int n = ng * TW + t * 16 + r;
    const float bv = bias ? bias[n] : 0.f;
    float* Cp = C + (size_t)(mtile * 16 + (half << 3)) * Nc + n;
    if (full) {
      if (Cbf) {
        unsigned short* Bp =
            Cbf + (size_t)(mtile * 16 + (half << 3)) * Nc + n;
#pragma unroll
        for (int i = 0; i < 8; ++i) {
          const float v = acc[t][i] + bv;
          Cp[(size_t)i * Nc] = v;
          Bp[(size_t)i * Nc] = f2bfbits(v);
        }
      } else {
#pragma unroll
        for (int i = 0; i < 8; ++i) Cp[(size_t)i * Nc] = acc[t][i] + bv;
      }
    } else {
#pragma unroll
      for (int i = 0; i < 8; ++i) {
        const int m = mtile * 16 + i + (half << 3);
        if (m < M) {
          const float v = acc[t][i] + bv;
          C[(size_t)m * Nc + n] = v;
          if (Cbf) Cbf[(size_t)m * Nc + n] = f2bfbits(v);
        }
      }
    }
  }
}

// ---------------------------------------------------------------------------
// Fills
// ---------------------------------------------------------------------------
__global__ void fill_f32(float* p, float v, long n) {
  long i = blockIdx.x * (long)blockDim.x + threadIdx.x;
  const long st = (long)gridDim.x * blockDim.x;
  for (; i < n; i += st) p[i] = v;
}
__global__ void fill_i32(int* p, int v, long n) {
  long i = blockIdx.x * (long)blockDim.x + threadIdx.x;
  const long st = (long)gridDim.x * blockDim.x;
  for (; i < n; i += st) p[i] = v;
}

// ---------------------------------------------------------------------------
// Pass 1: alpha_e = (Q[dst].K[src] + QWe[dst].ea[e]) / sqrt(128);
//         segment max via ordered-int atomicMax.  One wave per edge.
// ---------------------------------------------------------------------------
__global__ void edge_alpha_max(const float* __restrict__ Q,
                               const float* __restrict__ Kf,
                               const float* __restrict__ QWe,
                               const float* __restrict__ ea,
                               const int* __restrict__ src,
                               const int* __restrict__ dst,
                               float* __restrict__ alpha,
                               int* __restrict__ mEnc, int E_)
{
  const int e = (int)((blockIdx.x * blockDim.x + threadIdx.x) >> 5);
  const int lane = (int)(threadIdx.x & 31);
  if (e >= E_) return;
  const int s = src[e], d = dst[e];
  const float4 q4 = ((const float4*)(Q  + (size_t)d * 128))[lane];
  const float4 k4 = ((const float4*)(Kf + (size_t)s * 128))[lane];
  float p = q4.x * k4.x + q4.y * k4.y + q4.z * k4.z + q4.w * k4.w;
#pragma unroll
  for (int off = 16; off; off >>= 1) p += __shfl_xor(p, off, 32);
  if (lane == 0) {
    const float* qe = QWe + (size_t)d * 16;
    const float* ev = ea + (size_t)e * 16;
    float pe = 0.f;
#pragma unroll
    for (int j = 0; j < 16; ++j) pe += qe[j] * ev[j];
    const float a = (p + pe) * 0.08838834764831845f;   // 1/sqrt(128)
    alpha[e] = a;
    int enc = __float_as_int(a);
    enc = (enc >= 0) ? enc : (enc ^ 0x7FFFFFFF);       // ordered-int encode
    atomicMax(&mEnc[d], enc);
  }
}

__global__ void finalize_max(int* mEnc, int n) {
  const int i = (int)(blockIdx.x * blockDim.x + threadIdx.x);
  if (i >= n) return;
  const int e = mEnc[i];
  const int dec = (e >= 0) ? e : (e ^ 0x7FFFFFFF);
  float f = __int_as_float(dec);
  if (!isfinite(f)) f = 0.f;                           // empty segment -> 0
  ((float*)mEnc)[i] = f;
}

__global__ void edge_exp_sum(const float* __restrict__ alpha,
                             const float* __restrict__ m,
                             const int* __restrict__ dst,
                             float* __restrict__ aexp,
                             float* __restrict__ ssum, int E_)
{
  const int e = (int)(blockIdx.x * blockDim.x + threadIdx.x);
  if (e >= E_) return;
  const int d = dst[e];
  const float a = expf(alpha[e] - m[d]);
  aexp[e] = a;
  atomicAdd(&ssum[d], a);
}

__global__ void edge_aggregate(const float* __restrict__ aexp,
                               const float* __restrict__ ssum,
                               const float* __restrict__ V,
                               const float* __restrict__ ea,
                               const int* __restrict__ src,
                               const int* __restrict__ dst,
                               float* __restrict__ agg,
                               float* __restrict__ aggE, int E_)
{
  const int e = (int)((blockIdx.x * blockDim.x + threadIdx.x) >> 5);
  const int lane = (int)(threadIdx.x & 31);
  if (e >= E_) return;
  const int s = src[e], d = dst[e];
  const float w = aexp[e] / (ssum[d] + 1e-16f);
  const float4 v4 = ((const float4*)(V + (size_t)s * 128))[lane];
  float* ag = agg + (size_t)d * 128 + lane * 4;
  atomicAdd(ag + 0, w * v4.x);
  atomicAdd(ag + 1, w * v4.y);
  atomicAdd(ag + 2, w * v4.z);
  atomicAdd(ag + 3, w * v4.w);
  if (lane < 16)
    atomicAdd(&aggE[(size_t)d * 16 + lane], w * ea[(size_t)e * 16 + lane]);
}

// Epilogue: H' = gelu(agg + aggE@We + skip) (+ residual); writes f32 + bf16.
__global__ void epilogue(const float* __restrict__ agg,
                         const float* __restrict__ aggE,
                         const float* __restrict__ We,   // [16,128]
                         const float* __restrict__ S,
                         const float* __restrict__ Hres, // may be null
                         float* __restrict__ Hout,
                         unsigned short* __restrict__ Houtbf, int N_)
{
  const int n = (int)blockIdx.x;
  const int c = (int)threadIdx.x;            // 128 threads
  if (n >= N_) return;
  __shared__ float ae[16];
  if (c < 16) ae[c] = aggE[(size_t)n * 16 + c];
  __syncthreads();
  float acc = agg[(size_t)n * 128 + c] + S[(size_t)n * 128 + c];
#pragma unroll
  for (int j = 0; j < 16; ++j) acc += ae[j] * We[j * 128 + c];
  float g = 0.5f * acc * (1.f + erff(acc * 0.7071067811865476f));  // exact gelu
  if (Hres) g += Hres[(size_t)n * 128 + c];
  Hout[(size_t)n * 128 + c] = g;
  Houtbf[(size_t)n * 128 + c] = f2bfbits(g);
}

__global__ void pool_accum(const float* __restrict__ H,
                           const int* __restrict__ batch,
                           const int* __restrict__ mask,
                           float* __restrict__ pooled,
                           float* __restrict__ cnt, int N_)
{
  const int n = (int)blockIdx.x;
  const int c = (int)threadIdx.x;            // 128 threads
  if (n >= N_) return;
  if (mask[n]) {
    const int g = batch[n];
    atomicAdd(&pooled[(size_t)g * 128 + c], H[(size_t)n * 128 + c]);
    if (c == 0) atomicAdd(&cnt[g], 1.0f);
  }
}

__global__ void head(const float* __restrict__ pooled,
                     const float* __restrict__ cnt,
                     const float* __restrict__ W1, const float* __restrict__ b1,
                     const float* __restrict__ W2, const float* __restrict__ b2,
                     float* __restrict__ out)
{
  const int g = (int)blockIdx.x;
  const int c = (int)threadIdx.x;            // 128 threads
  __shared__ float pm[128];
  __shared__ float hid[128];
  float cc = cnt[g];
  cc = cc > 1.f ? cc : 1.f;
  pm[c] = pooled[(size_t)g * 128 + c] / cc;
  __syncthreads();
  float a = b1[c];
  for (int k = 0; k < 128; ++k) a += pm[k] * W1[k * 128 + c];
  hid[c] = a > 0.f ? a : 0.f;
  __syncthreads();
  if (c < 3) {
    float o = b2[c];
    for (int k = 0; k < 128; ++k) o += hid[k] * W2[k * 3 + c];
    out[(size_t)g * 3 + c] = o;
  }
}

// ---------------------------------------------------------------------------
extern "C" void kernel_launch(void* const* d_in, const int* in_sizes, int n_in,
                              void* d_out, int out_size, void* d_ws, size_t ws_size,
                              hipStream_t stream)
{
  const int F_IN = 32, D = 128, EDIM = 16;
  const int N = in_sizes[0] / F_IN;        // 50000
  const int E = in_sizes[1] / EDIM;        // 600000
  const int L = in_sizes[11] / (D * D);    // 3
  const int G = out_size / 3;              // 64

  const float* x     = (const float*)d_in[0];
  const float* eattr = (const float*)d_in[1];
  const float* c1_Wq = (const float*)d_in[2];
  const float* c1_bq = (const float*)d_in[3];
  const float* c1_Wk = (const float*)d_in[4];
  const float* c1_bk = (const float*)d_in[5];
  const float* c1_Wv = (const float*)d_in[6];
  const float* c1_bv = (const float*)d_in[7];
  const float* c1_We = (const float*)d_in[8];
  const float* c1_Ws = (const float*)d_in[9];
  const float* c1_bs = (const float*)d_in[10];
  const float* cv_Wq = (const float*)d_in[11];
  const float* cv_bq = (const float*)d_in[12];
  const float* cv_Wk = (const float*)d_in[13];
  const float* cv_bk = (const float*)d_in[14];
  const float* cv_Wv = (const float*)d_in[15];
  const float* cv_bv = (const float*)d_in[16];
  const float* cv_We = (const float*)d_in[17];
  const float* cv_Ws = (const float*)d_in[18];
  const float* cv_bs = (const float*)d_in[19];
  const float* linl_W  = (const float*)d_in[20];
  const float* linl_b  = (const float*)d_in[21];
  const float* linl2_W = (const float*)d_in[22];
  const float* linl2_b = (const float*)d_in[23];
  const int* eidx  = (const int*)d_in[24];
  const int* batch = (const int*)d_in[25];
  const int* flex  = (const int*)d_in[26];
  const int* esrc = eidx;
  const int* edst = eidx + E;
  float* out = (float*)d_out;

  // ---- workspace carve ----
  size_t off = 0;
  auto carve = [&](size_t bytes) -> char* {
    char* p = (char*)d_ws + off;
    off += (bytes + 255) & ~(size_t)255;
    return p;
  };
  float* Q    = (float*)carve((size_t)N * D * 4);
  float* Kf   = (float*)carve((size_t)N * D * 4);
  float* Vf   = (float*)carve((size_t)N * D * 4);
  float* Sf   = (float*)carve((size_t)N * D * 4);
  float* QWe  = (float*)carve((size_t)N * EDIM * 4);
  float* HA   = (float*)carve((size_t)N * D * 4);
  float* HB   = (float*)carve((size_t)N * D * 4);
  float* agg  = (float*)carve((size_t)N * D * 4);
  float* aggE = (float*)carve((size_t)N * EDIM * 4);
  float* alpha = (float*)carve((size_t)E * 4);
  float* aexp  = (float*)carve((size_t)E * 4);
  int*   mbuf  = (int*)carve((size_t)N * 4);
  float* ssum  = (float*)carve((size_t)N * 4);
  float* pooled = (float*)carve((size_t)G * D * 4);
  float* cnt    = (float*)carve((size_t)G * 4);
  unsigned short* xbf = (unsigned short*)carve((size_t)N * F_IN * 2);
  unsigned short* Hbf = (unsigned short*)carve((size_t)N * D * 2);
  unsigned short* Qbf = (unsigned short*)carve((size_t)N * D * 2);
  unsigned int* Pq = (unsigned int*)carve((size_t)(D / 2) * D * 4);
  unsigned int* Pk = (unsigned int*)carve((size_t)(D / 2) * D * 4);
  unsigned int* Pv = (unsigned int*)carve((size_t)(D / 2) * D * 4);
  unsigned int* Ps = (unsigned int*)carve((size_t)(D / 2) * D * 4);
  unsigned int* Pe = (unsigned int*)carve((size_t)(D / 2) * EDIM * 4);
  (void)ws_size; (void)n_in;

  const int mgroups = ((N + 15) / 16 + 7) / 8;           // 8 waves / block
  // Dispatch: (K, Nc) -> template instantiation (NT=2 when Nc allows)
  auto gemm = [&](const unsigned short* Abf, int K, const unsigned int* Bpk,
                  int Nc, const float* bias, float* C, unsigned short* Cbf) {
    if (Nc >= 32) {
      dim3 grid(mgroups, Nc / 32);
      if (K == 32)
        wmma_gemm_bf16<32, 2><<<grid, 256, 0, stream>>>(Abf, Bpk, Nc, bias, C,
                                                        Cbf, N);
      else
        wmma_gemm_bf16<128, 2><<<grid, 256, 0, stream>>>(Abf, Bpk, Nc, bias, C,
                                                         Cbf, N);
    } else {
      dim3 grid(mgroups, Nc / 16);
      wmma_gemm_bf16<128, 1><<<grid, 256, 0, stream>>>(Abf, Bpk, Nc, bias, C,
                                                       Cbf, N);
    }
  };
  auto packW = [&](const float* W, int sK, int sN, unsigned int* Bpk,
                   int K, int Nc) {
    const int total = (K / 2) * Nc;
    pack_B<<<(total + 255) / 256, 256, 0, stream>>>(W, sK, sN, Bpk, K, Nc);
  };

  const int eWaveBlocks = (E + 7) / 8;
  const int eThrBlocks  = (E + 255) / 256;
  const int nThrBlocks  = (N + 255) / 256;

  auto run_conv = [&](const unsigned short* Abf, int din,
                      const float* Wq, const float* bq,
                      const float* Wk, const float* bk,
                      const float* Wv, const float* bv,
                      const float* We,
                      const float* Ws, const float* bs,
                      const float* Hres, float* Hout) {
    packW(Wq, D, 1, Pq, din, D);
    packW(Wk, D, 1, Pk, din, D);
    packW(Wv, D, 1, Pv, din, D);
    packW(Ws, D, 1, Ps, din, D);
    packW(We, 1, D, Pe, D, EDIM);          // B = We^T : B[k,n] = We[n*128+k]

    gemm(Abf, din, Pq, D, bq, Q, Qbf);
    gemm(Abf, din, Pk, D, bk, Kf, nullptr);
    gemm(Abf, din, Pv, D, bv, Vf, nullptr);
    gemm(Abf, din, Ps, D, bs, Sf, nullptr);
    gemm(Qbf, D, Pe, EDIM, nullptr, QWe, nullptr);       // QWe = Q @ We^T

    fill_i32<<<1024, 256, 0, stream>>>(mbuf, (int)0x807FFFFF, N);   // enc(-inf)
    fill_f32<<<1024, 256, 0, stream>>>(ssum, 0.f, N);
    fill_f32<<<4096, 256, 0, stream>>>(agg, 0.f, (long)N * D);
    fill_f32<<<1024, 256, 0, stream>>>(aggE, 0.f, (long)N * EDIM);

    edge_alpha_max<<<eWaveBlocks, 256, 0, stream>>>(Q, Kf, QWe, eattr, esrc,
                                                    edst, alpha, mbuf, E);
    finalize_max<<<nThrBlocks, 256, 0, stream>>>(mbuf, N);
    edge_exp_sum<<<eThrBlocks, 256, 0, stream>>>(alpha, (const float*)mbuf,
                                                 edst, aexp, ssum, E);
    edge_aggregate<<<eWaveBlocks, 256, 0, stream>>>(aexp, ssum, Vf, eattr,
                                                    esrc, edst, agg, aggE, E);
    epilogue<<<N, 128, 0, stream>>>(agg, aggE, We, Sf, Hres, Hout, Hbf, N);
  };

  // conv1: x[N,32] -> HA (+ Hbf), gelu, no residual
  cvt_f32_bf16<<<1024, 256, 0, stream>>>(x, xbf, (long)N * F_IN);
  run_conv(xbf, F_IN, c1_Wq, c1_bq, c1_Wk, c1_bk, c1_Wv, c1_bv, c1_We,
           c1_Ws, c1_bs, nullptr, HA);

  // residual layers (layer i reads Hbf written by previous epilogue)
  float* cur = HA;
  float* nxt = HB;
  for (int i = 0; i < L; ++i) {
    run_conv(Hbf, D,
             cv_Wq + (size_t)i * D * D, cv_bq + (size_t)i * D,
             cv_Wk + (size_t)i * D * D, cv_bk + (size_t)i * D,
             cv_Wv + (size_t)i * D * D, cv_bv + (size_t)i * D,
             cv_We + (size_t)i * EDIM * D,
             cv_Ws + (size_t)i * D * D, cv_bs + (size_t)i * D,
             cur, nxt);
    float* t = cur; cur = nxt; nxt = t;
  }

  // masked mean pool + MLP head
  fill_f32<<<64, 256, 0, stream>>>(pooled, 0.f, (long)G * D);
  fill_f32<<<1, 64, 0, stream>>>(cnt, 0.f, G);
  pool_accum<<<N, 128, 0, stream>>>(cur, batch, flex, pooled, cnt, N);
  head<<<G, 128, 0, stream>>>(pooled, cnt, linl_W, linl_b, linl2_W, linl2_b, out);
}
